// HydraTrajHead_3023656976419
// MI455X (gfx1250) — matrix operand order repaired and checked by reading, not compiled, not run on hardware
//
#include <hip/hip_runtime.h>
#include <hip/hip_bf16.h>
#include <math.h>

typedef _Float16 h16;
typedef __attribute__((ext_vector_type(16))) _Float16 v16h;
typedef __attribute__((ext_vector_type(8)))  _Float16 v8h;
typedef __attribute__((ext_vector_type(8)))  float    v8f;

#define B_    4
#define L_    2048
#define T_    40
#define S_    512
#define D_    512
#define F_    2048
#define H_    8
#define DH_   64
#define TOPK_ 256
#define BL    (B_*L_)          // 8192 rows

// Async global->LDS copy, 16B per lane, per-lane LDS destination (ASYNCcnt).
#define ASYNC_B128(gp, loff)                                            \
  asm volatile("global_load_async_to_lds_b128 %0, %1, off"              \
               :: "v"(loff), "v"(gp) : "memory")
#define WAIT_ASYNC_4() asm volatile("s_wait_asynccnt 4" ::: "memory")
#define WAIT_ASYNC_0() asm volatile("s_wait_asynccnt 0" ::: "memory")

__device__ __forceinline__ unsigned lds_addr(const h16* p) {
  return (unsigned)(uintptr_t)p;   // generic LDS address: low 32 bits = LDS offset
}

// ---------------------------------------------------------------------------
// f32 -> f16 convert with optional zero padding (A-side tensors)
// ---------------------------------------------------------------------------
__global__ void k_cvt_pad(const float* __restrict__ src, h16* __restrict__ dst,
                          int srcRows, int srcCols, int dstCols, int n) {
  int i = blockIdx.x * 256 + threadIdx.x;
  if (i >= n) return;
  int r = i / dstCols, c = i - r * dstCols;
  float v = (r < srcRows && c < srcCols) ? src[(size_t)r * srcCols + c] : 0.f;
  dst[i] = (h16)v;
}

// dst[N, Kpad] f16 row-major = transpose(src[Ksrc, N] f32), zero pad k>=Ksrc.
__global__ void k_cvt_tr(const float* __restrict__ src, h16* __restrict__ dst,
                         int Ksrc, int N, int Kpad, int n) {
  int i = blockIdx.x * 256 + threadIdx.x;
  if (i >= n) return;
  int nn = i / Kpad, k = i - nn * Kpad;
  float v = (k < Ksrc) ? src[(size_t)k * N + nn] : 0.f;
  dst[i] = (h16)v;
}

// ---------------------------------------------------------------------------
// WMMA GEMM: C[M,N] = act(A[M,K](f16) @ W[K,N] + bias), f32 accumulate.
// W supplied pre-transposed as Bt[N,K]. Block tile 128x128, K-step 32,
// 8 waves (4M x 2N), each wave 32x64 of C; double-buffered async staging.
// ---------------------------------------------------------------------------
#define BM 128
#define BN 128
#define BKK 32
#define LDS_A 40
#define LDS_B 40
#define A_BUF (BM*LDS_A)
#define B_BUF (BN*LDS_B)

__global__ __launch_bounds__(256) void k_gemm(
    const h16* __restrict__ A, int lda,
    const h16* __restrict__ Bt, int ldb,
    const float* __restrict__ bias,
    float* __restrict__ out32, h16* __restrict__ out16, int ldc,
    int K, int relu)
{
  __shared__ h16 As[2 * A_BUF];
  __shared__ h16 Bs[2 * B_BUF];

  const int tid  = threadIdx.x;
  const int lane = tid & 31;
  const int wave = tid >> 5;
  const int wm   = wave & 3;
  const int wn   = wave >> 2;
  const int mBase = blockIdx.y * BM;
  const int nBase = blockIdx.x * BN;

  const int g0 = tid, g1 = tid + 256;
  const int r0 = g0 >> 2, c0 = (g0 & 3) * 8;
  const int r1 = g1 >> 2, c1 = (g1 & 3) * 8;
  const h16* gA0 = A  + (size_t)(mBase + r0) * lda + c0;
  const h16* gA1 = A  + (size_t)(mBase + r1) * lda + c1;
  const h16* gB0 = Bt + (size_t)(nBase + r0) * ldb + c0;
  const h16* gB1 = Bt + (size_t)(nBase + r1) * ldb + c1;
  unsigned lA0[2], lA1[2], lB0[2], lB1[2];
#pragma unroll
  for (int b = 0; b < 2; ++b) {
    lA0[b] = lds_addr(&As[b * A_BUF + r0 * LDS_A + c0]);
    lA1[b] = lds_addr(&As[b * A_BUF + r1 * LDS_A + c1]);
    lB0[b] = lds_addr(&Bs[b * B_BUF + r0 * LDS_B + c0]);
    lB1[b] = lds_addr(&Bs[b * B_BUF + r1 * LDS_B + c1]);
  }
#define GEMM_ISSUE(buf, k0) do {            \
    ASYNC_B128(gA0 + (k0), lA0[buf]);       \
    ASYNC_B128(gA1 + (k0), lA1[buf]);       \
    ASYNC_B128(gB0 + (k0), lB0[buf]);       \
    ASYNC_B128(gB1 + (k0), lB1[buf]); } while (0)

  v8f acc[2][4] = {};

  const int rsel = lane & 15;
  const int kb8  = (lane >> 4) << 3;
  const int kb16 = (lane >> 4) << 4;

  const int nT = K / BKK;
  GEMM_ISSUE(0, 0);
  for (int t = 0; t < nT; ++t) {
    if (t + 1 < nT) { GEMM_ISSUE((t + 1) & 1, (t + 1) * BKK); WAIT_ASYNC_4(); }
    else            { WAIT_ASYNC_0(); }
    __syncthreads();

    const int ab = (t & 1) * A_BUF;
    const int bb = (t & 1) * B_BUF;
    v16h afrag[2], bfrag[4];
#pragma unroll
    for (int i = 0; i < 2; ++i) {
      const h16* p = &As[ab + (wm * 32 + i * 16 + rsel) * LDS_A];
      v8h lo = *(const v8h*)(p + kb16);
      v8h hi = *(const v8h*)(p + kb16 + 8);
#pragma unroll
      for (int q = 0; q < 8; ++q) { afrag[i][q] = lo[q]; afrag[i][q + 8] = hi[q]; }
    }
#pragma unroll
    for (int j = 0; j < 4; ++j) {
      const h16* p = &Bs[bb + (wn * 64 + j * 16 + rsel) * LDS_B + kb16];
      v8h lo = *(const v8h*)(p);
      v8h hi = *(const v8h*)(p + 8);
#pragma unroll
      for (int q = 0; q < 8; ++q) { bfrag[j][q] = lo[q]; bfrag[j][q + 8] = hi[q]; }
    }
#pragma unroll
    for (int i = 0; i < 2; ++i)
#pragma unroll
      for (int j = 0; j < 4; ++j)
        acc[i][j] = __builtin_amdgcn_wmma_f32_16x16x32_f16(
            false, afrag[i], false, bfrag[j], (short)0, acc[i][j], false, false);
    __syncthreads();
  }

  const int rowOff = (lane >> 4) << 3;
  const int colL   = lane & 15;
#pragma unroll
  for (int i = 0; i < 2; ++i)
#pragma unroll
    for (int j = 0; j < 4; ++j)
#pragma unroll
      for (int v = 0; v < 8; ++v) {
        int row = mBase + wm * 32 + i * 16 + v + rowOff;
        int col = nBase + wn * 64 + j * 16 + colL;
        float x = acc[i][j][v] + (bias ? bias[col] : 0.f);
        if (relu) x = fmaxf(x, 0.f);
        size_t o = (size_t)row * ldc + col;
        if (out32) out32[o] = x;
        if (out16) out16[o] = (h16)x;
      }
#undef GEMM_ISSUE
}

// ---------------------------------------------------------------------------
// V transpose: Vt[(b*H+h)*64 + d][key] = V[(b*Lk + key)*D + h*64 + d]
// LDS-tiled 64x64 so both global accesses are b128-coalesced.
// ---------------------------------------------------------------------------
__global__ __launch_bounds__(256) void k_vtr(const h16* __restrict__ V,
                                             h16* __restrict__ Vt, int Lk) {
  __shared__ h16 t[64 * 72];
  const int bh = blockIdx.y, b = bh >> 3, h = bh & 7;
  const int k0 = blockIdx.x * 64;
  const int tid = threadIdx.x;
#pragma unroll
  for (int i = 0; i < 2; ++i) {
    int g = i * 256 + tid;
    int key = g >> 3, d = (g & 7) * 8;
    *(float4*)&t[key * 72 + d] =
        *(const float4*)(V + ((size_t)b * Lk + k0 + key) * D_ + h * DH_ + d);
  }
  __syncthreads();
#pragma unroll
  for (int i = 0; i < 2; ++i) {
    int g = i * 256 + tid;
    int d = g >> 3, key = (g & 7) * 8;
    union { float4 f; h16 h[8]; } u;
#pragma unroll
    for (int q = 0; q < 8; ++q) u.h[q] = t[(key + q) * 72 + d];
    *(float4*)(Vt + ((size_t)bh * DH_ + d) * Lk + k0 + key) = u.f;
  }
}

// ---------------------------------------------------------------------------
// WMMA flash attention. One (b,h) per blockIdx.y; 8 waves x 16 q-rows each.
// K staged [key][d] (natural) and V staged [d][key] (pre-transposed) in
// double-buffered async LDS chunks of 64 keys. Per chunk per wave:
// 8 WMMA (QK^T) + online softmax on C fragments (shfl row reductions) +
// P transpose via wave-private LDS tile + 8 WMMA (P V).
// ---------------------------------------------------------------------------
#define AKV (64 * 72)

__global__ __launch_bounds__(256) void k_attn_wmma(
    const h16* __restrict__ Q, const h16* __restrict__ Kt,
    const h16* __restrict__ Vt, h16* __restrict__ O, int Lk)
{
  __shared__ h16 Ks[2 * AKV];        // [key][d], stride 72
  __shared__ h16 Vs[2 * AKV];        // [d][key], stride 72
  __shared__ h16 Pw[8 * 16 * 72];    // per-wave P tile [row][key]

  const int tid  = threadIdx.x, lane = tid & 31, wave = tid >> 5;
  const int bh   = blockIdx.y, b = bh >> 3, h = bh & 7;
  const int qBase = blockIdx.x * 128 + wave * 16;
  const int rsel = lane & 15, hi = lane >> 4;
  const int kb8  = hi * 8, kb16 = hi * 16;

  // Q fragments, softmax scale 1/8 folded in (exact in f16)
  v16h aQ[2];
  {
    const h16* qp = Q + ((size_t)b * L_ + qBase + rsel) * D_ + h * DH_;
#pragma unroll
    for (int f = 0; f < 2; ++f) {
      v8h lo = *(const v8h*)(qp + f * 32 + kb16);
      v8h hp = *(const v8h*)(qp + f * 32 + kb16 + 8);
#pragma unroll
      for (int q = 0; q < 8; ++q) {
        aQ[f][q]     = lo[q] * (h16)0.125f;
        aQ[f][q + 8] = hp[q] * (h16)0.125f;
      }
    }
  }

  const h16* kbp = Kt + ((size_t)b * Lk) * D_ + h * DH_;
  const h16* vtp = Vt + ((size_t)bh * DH_) * Lk;
  const h16* gK[2]; const h16* gV[2];
  unsigned lK[2][2], lV[2][2];
#pragma unroll
  for (int t = 0; t < 2; ++t) {
    int g = t * 256 + tid;
    int row = g >> 3, col = (g & 7) * 8;
    gK[t] = kbp + (size_t)row * D_ + col;
    gV[t] = vtp + (size_t)row * Lk + col;
#pragma unroll
    for (int bf = 0; bf < 2; ++bf) {
      lK[bf][t] = lds_addr(&Ks[bf * AKV + row * 72 + col]);
      lV[bf][t] = lds_addr(&Vs[bf * AKV + row * 72 + col]);
    }
  }
#define AT_ISSUE(buf, kc) do {                            \
    ASYNC_B128(gK[0] + (size_t)(kc) * D_, lK[buf][0]);    \
    ASYNC_B128(gK[1] + (size_t)(kc) * D_, lK[buf][1]);    \
    ASYNC_B128(gV[0] + (kc),              lV[buf][0]);    \
    ASYNC_B128(gV[1] + (kc),              lV[buf][1]); } while (0)

  float m[8], l[8];
  v8f oAcc[4] = {};
#pragma unroll
  for (int v = 0; v < 8; ++v) { m[v] = -1e30f; l[v] = 0.f; }

  h16* pw = &Pw[wave * 16 * 72];

  AT_ISSUE(0, 0);
  for (int kc = 0; kc < Lk; kc += 64) {
    if (kc + 64 < Lk) { AT_ISSUE(((kc >> 6) + 1) & 1, kc + 64); WAIT_ASYNC_4(); }
    else              { WAIT_ASYNC_0(); }
    __syncthreads();
    const int cb = ((kc >> 6) & 1) * AKV;

    // ---- S = Q K^T over 4 key blocks of 16 ----
    v8f sf[4];
#pragma unroll
    for (int kb = 0; kb < 4; ++kb) {
      const h16* kp = &Ks[cb + (kb * 16 + rsel) * 72];
      v16h b0, b1;
      v8h x0 = *(const v8h*)(kp + kb16);
      v8h x1 = *(const v8h*)(kp + kb16 + 8);
      v8h x2 = *(const v8h*)(kp + 32 + kb16);
      v8h x3 = *(const v8h*)(kp + 32 + kb16 + 8);
#pragma unroll
      for (int q = 0; q < 8; ++q) { b0[q] = x0[q]; b0[q + 8] = x1[q];
                                    b1[q] = x2[q]; b1[q + 8] = x3[q]; }
      v8f s = {};
      s = __builtin_amdgcn_wmma_f32_16x16x32_f16(false, aQ[0], false, b0, (short)0, s, false, false);
      s = __builtin_amdgcn_wmma_f32_16x16x32_f16(false, aQ[1], false, b1, (short)0, s, false, false);
      sf[kb] = s;
    }

    // ---- online softmax: row stats via shfl within 16-lane halves ----
#pragma unroll
    for (int v = 0; v < 8; ++v) {
      float rmax = fmaxf(fmaxf(sf[0][v], sf[1][v]), fmaxf(sf[2][v], sf[3][v]));
#pragma unroll
      for (int msk = 1; msk < 16; msk <<= 1)
        rmax = fmaxf(rmax, __shfl_xor(rmax, msk, 32));
      float mn = fmaxf(m[v], rmax);
      float cs = __expf(m[v] - mn);
      m[v] = mn; l[v] *= cs;
      oAcc[0][v] *= cs; oAcc[1][v] *= cs; oAcc[2][v] *= cs; oAcc[3][v] *= cs;
    }

    // ---- P = exp(S - m): store f16 to wave-private LDS, row sums ----
    float ps[8] = {};
#pragma unroll
    for (int kb = 0; kb < 4; ++kb)
#pragma unroll
      for (int v = 0; v < 8; ++v) {
        float p = __expf(sf[kb][v] - m[v]);
        ps[v] += p;
        pw[(v + 8 * hi) * 72 + kb * 16 + rsel] = (h16)p;
      }
#pragma unroll
    for (int v = 0; v < 8; ++v) {
      float t = ps[v];
#pragma unroll
      for (int msk = 1; msk < 16; msk <<= 1) t += __shfl_xor(t, msk, 32);
      l[v] += t;
    }

    // ---- read P back as A fragments (same-wave LDS is in-order) ----
    v16h pA[2];
#pragma unroll
    for (int f = 0; f < 2; ++f) {
      const h16* pp = &pw[rsel * 72 + f * 32];
      v8h lo = *(const v8h*)(pp + kb16);
      v8h hp = *(const v8h*)(pp + kb16 + 8);
#pragma unroll
      for (int q = 0; q < 8; ++q) { pA[f][q] = lo[q]; pA[f][q + 8] = hp[q]; }
    }

    // ---- O += P V ----
#pragma unroll
    for (int db = 0; db < 4; ++db) {
      const h16* vp = &Vs[cb + (db * 16 + rsel) * 72];
      v16h b0, b1;
      v8h y0 = *(const v8h*)(vp + kb16);
      v8h y1 = *(const v8h*)(vp + kb16 + 8);
      v8h y2 = *(const v8h*)(vp + 32 + kb16);
      v8h y3 = *(const v8h*)(vp + 32 + kb16 + 8);
#pragma unroll
      for (int q = 0; q < 8; ++q) { b0[q] = y0[q]; b0[q + 8] = y1[q];
                                    b1[q] = y2[q]; b1[q + 8] = y3[q]; }
      oAcc[db] = __builtin_amdgcn_wmma_f32_16x16x32_f16(false, pA[0], false, b0, (short)0, oAcc[db], false, false);
      oAcc[db] = __builtin_amdgcn_wmma_f32_16x16x32_f16(false, pA[1], false, b1, (short)0, oAcc[db], false, false);
    }
    __syncthreads();
  }

  // ---- normalize and write O (f16, feeds O-projection GEMM) ----
  h16* op = O + ((size_t)b * L_ + qBase) * D_ + h * DH_;
#pragma unroll
  for (int v = 0; v < 8; ++v) {
    float inv = 1.f / l[v];
    int row = v + 8 * hi;
#pragma unroll
    for (int db = 0; db < 4; ++db)
      op[(size_t)row * D_ + db * 16 + rsel] = (h16)(oAcc[db][v] * inv);
  }
#undef AT_ISSUE
}

// ---------------------------------------------------------------------------
// LayerNorm over D=512: x = LN(x + res); writes f32 (in-place) + f16 mirror.
// ---------------------------------------------------------------------------
__global__ __launch_bounds__(128) void k_ln(
    float* __restrict__ x, const float* __restrict__ res,
    const float* __restrict__ g, const float* __restrict__ bb,
    h16* __restrict__ x16)
{
  __shared__ float red[128];
  const int row = blockIdx.x, tid = threadIdx.x;
  float v[4], s = 0.f;
#pragma unroll
  for (int t = 0; t < 4; ++t) {
    int c = tid + t * 128;
    float a = x[(size_t)row * D_ + c] + res[(size_t)row * D_ + c];
    v[t] = a; s += a;
  }
  red[tid] = s; __syncthreads();
  for (int st = 64; st > 0; st >>= 1) { if (tid < st) red[tid] += red[tid + st]; __syncthreads(); }
  float mean = red[0] * (1.f / D_); __syncthreads();
  float s2 = 0.f;
#pragma unroll
  for (int t = 0; t < 4; ++t) { float d = v[t] - mean; s2 += d * d; }
  red[tid] = s2; __syncthreads();
  for (int st = 64; st > 0; st >>= 1) { if (tid < st) red[tid] += red[tid + st]; __syncthreads(); }
  float inv = rsqrtf(red[0] * (1.f / D_) + 1e-5f);
#pragma unroll
  for (int t = 0; t < 4; ++t) {
    int c = tid + t * 128;
    float y = g[c] * (v[t] - mean) * inv + bb[c];
    x[(size_t)row * D_ + c] = y;
    x16[(size_t)row * D_ + c] = (h16)y;
  }
}

// broadcast pos-embed [L,D] f32 to x[B,L,D] f32+f16
__global__ void k_bcast(const float* __restrict__ src, float* __restrict__ x,
                        h16* __restrict__ x16, int n) {
  int i = blockIdx.x * 256 + threadIdx.x;
  if (i >= n) return;
  float v = src[i & (L_ * D_ - 1)];
  x[i] = v; x16[i] = (h16)v;
}

// x += status[b] broadcast; refresh f16 mirror
__global__ void k_addstatus(float* __restrict__ x, const float* __restrict__ st,
                            h16* __restrict__ x16) {
  int i = blockIdx.x * 256 + threadIdx.x;
  if (i >= BL * D_) return;
  int b = i >> 20;
  int d = i & (D_ - 1);
  float v = x[i] + st[b * D_ + d];
  x[i] = v; x16[i] = (h16)v;
}

// matvec: out[m] = hidden[m,:K](f16) . w[:K](f32) + bias[0]; wave per row
__global__ __launch_bounds__(256) void k_matvec(
    const h16* __restrict__ Hd, const float* __restrict__ w,
    const float* __restrict__ bias, float* __restrict__ out, int K)
{
  const int lane = threadIdx.x & 31, wave = threadIdx.x >> 5;
  const int row = blockIdx.x * 8 + wave;
  const h16* hp = Hd + (size_t)row * K;
  float s = 0.f;
  for (int k = lane; k < K; k += 32) s = fmaf((float)hp[k], w[k], s);
  for (int off = 16; off; off >>= 1) s += __shfl_down(s, off, 32);
  if (lane == 0) out[row] = s + bias[0];
}

__device__ __forceinline__ float d_sig(float x)  { return 1.f / (1.f + __expf(-x)); }
__device__ __forceinline__ float d_lsig(float x) { return fminf(x, 0.f) - log1pf(__expf(-fabsf(x))); }

// rbuf slots: 0 dac, 1 ddc, 2 ep, 3 lk, 4 nafc, 5 ttc, 6 tlc, 7 rl, 8 rl_topk
__global__ __launch_bounds__(256) void k_scores(const float* __restrict__ rb,
                                                float* __restrict__ scores) {
  __shared__ float red[256];
  const int b = blockIdx.x, tid = threadIdx.x;
  const float* rl = rb + 7 * BL + b * L_;
  float mx = -1e30f;
  for (int i = tid; i < L_; i += 256) mx = fmaxf(mx, rl[i]);
  red[tid] = mx; __syncthreads();
  for (int s = 128; s; s >>= 1) { if (tid < s) red[tid] = fmaxf(red[tid], red[tid + s]); __syncthreads(); }
  mx = red[0]; __syncthreads();
  float se = 0.f;
  for (int i = tid; i < L_; i += 256) se += __expf(rl[i] - mx);
  red[tid] = se; __syncthreads();
  for (int s = 128; s; s >>= 1) { if (tid < s) red[tid] += red[tid + s]; __syncthreads(); }
  float lse = mx + __logf(red[0]);
  for (int i = tid; i < L_; i += 256) {
    int gi = b * L_ + i;
    float v = 0.05f * (rl[i] - lse)
      + 0.5f * (d_lsig(rb[6 * BL + gi]) + d_lsig(rb[4 * BL + gi]) +
                d_lsig(rb[0 * BL + gi]) + d_lsig(rb[1 * BL + gi]))
      + 8.f * __logf(5.f * d_sig(rb[5 * BL + gi]) + 5.f * d_sig(rb[2 * BL + gi]) +
                     5.f * d_sig(rb[3 * BL + gi]));
    scores[gi] = v;
  }
}

// top-256 selection + log_softmax(r_topk[idx]) scatter-add (order-invariant)
__global__ __launch_bounds__(256) void k_topk(float* __restrict__ scores,
                                              const float* __restrict__ rtopk) {
  __shared__ float sc[L_];
  __shared__ float rv[256];
  __shared__ int   ri[256];
  __shared__ int   sel[TOPK_];
  const int b = blockIdx.x, tid = threadIdx.x;
  for (int i = tid; i < L_; i += 256) sc[i] = scores[b * L_ + i];
  __syncthreads();
  for (int it = 0; it < TOPK_; ++it) {
    float bv = -1e30f; int bi = L_;
    for (int i = tid; i < L_; i += 256) {
      float v = sc[i];
      if (v > bv || (v == bv && i < bi)) { bv = v; bi = i; }
    }
    rv[tid] = bv; ri[tid] = bi; __syncthreads();
    for (int s = 128; s; s >>= 1) {
      if (tid < s && (rv[tid + s] > rv[tid] || (rv[tid + s] == rv[tid] && ri[tid + s] < ri[tid]))) {
        rv[tid] = rv[tid + s]; ri[tid] = ri[tid + s];
      }
      __syncthreads();
    }
    if (tid == 0) { sel[it] = ri[0]; sc[ri[0]] = -1e30f; }
    __syncthreads();
  }
  const int idx = sel[tid];
  const float v = rtopk[b * L_ + idx];
  rv[tid] = v; __syncthreads();
  for (int s = 128; s; s >>= 1) { if (tid < s) rv[tid] = fmaxf(rv[tid], rv[tid + s]); __syncthreads(); }
  float mx = rv[0]; __syncthreads();
  rv[tid] = __expf(v - mx); __syncthreads();
  for (int s = 128; s; s >>= 1) { if (tid < s) rv[tid] += rv[tid + s]; __syncthreads(); }
  float lse = mx + __logf(rv[0]);
  scores[b * L_ + idx] += 0.01f * (v - lse);
}

// final argmax + outputs: scores[8192] | idx[4] (int bits) | traj[4*40*3]
__global__ __launch_bounds__(256) void k_output(const float* __restrict__ scores,
                                                const float* __restrict__ vocab,
                                                float* __restrict__ out) {
  __shared__ float rv[256];
  __shared__ int   ri[256];
  const int b = blockIdx.x, tid = threadIdx.x;
  float bv = -1e30f; int bi = L_;
  for (int i = tid; i < L_; i += 256) {
    float v = scores[b * L_ + i];
    if (v > bv || (v == bv && i < bi)) { bv = v; bi = i; }
  }
  rv[tid] = bv; ri[tid] = bi; __syncthreads();
  for (int s = 128; s; s >>= 1) {
    if (tid < s && (rv[tid + s] > rv[tid] || (rv[tid + s] == rv[tid] && ri[tid + s] < ri[tid]))) {
      rv[tid] = rv[tid + s]; ri[tid] = ri[tid + s];
    }
    __syncthreads();
  }
  const int sel = ri[0];
  for (int i = tid; i < L_; i += 256) out[b * L_ + i] = scores[b * L_ + i];
  if (tid == 0) ((int*)out)[BL + b] = sel;
  if (tid < T_ * 3) out[BL + B_ + b * T_ * 3 + tid] = vocab[(size_t)sel * T_ * 3 + tid];
}

// ---------------------------------------------------------------------------
// host orchestration
// ---------------------------------------------------------------------------
extern "C" void kernel_launch(void* const* d_in, const int* in_sizes, int n_in,
                              void* d_out, int out_size, void* d_ws, size_t ws_size,
                              hipStream_t stream) {
  (void)in_sizes; (void)out_size;
  if (n_in < 103) return;

  const float* bev    = (const float*)d_in[0];
  const float* status = (const float*)d_in[1];
  const float* vocab  = (const float*)d_in[2];

  char* ws = (char*)d_ws;
  size_t off = 0;
  auto alloc = [&](size_t bytes) -> void* {
    void* p = ws + off; off = (off + bytes + 255) & ~(size_t)255; return p;
  };
  h16*   wcvt   = (h16*)  alloc((size_t)F_ * F_ * 2);   // transposed weight slab
  float* xf32   = (float*)alloc((size_t)BL * D_ * 4);
  h16*   xf16   = (h16*)  alloc((size_t)BL * D_ * 2);
  float* tmp32  = (float*)alloc((size_t)BL * D_ * 4);
  h16*   q16    = (h16*)  alloc((size_t)BL * D_ * 2);
  h16*   k16    = (h16*)  alloc((size_t)BL * D_ * 2);
  h16*   v16    = (h16*)  alloc((size_t)BL * D_ * 2);
  h16*   v16t   = (h16*)  alloc((size_t)B_ * H_ * DH_ * L_ * 2);
  h16*   ao16   = (h16*)  alloc((size_t)BL * D_ * 2);
  h16*   bigh   = (h16*)  alloc((size_t)BL * F_ * 2);
  h16*   bigh2  = (h16*)  alloc((size_t)BL * F_ * 2);
  h16*   bev16  = (h16*)  alloc((size_t)B_ * S_ * D_ * 2);
  h16*   vpad   = (h16*)  alloc((size_t)L_ * 128 * 2);
  float* rbuf   = (float*)alloc((size_t)9 * BL * 4);
  float* scores = (float*)alloc((size_t)BL * 4);
  if (off > ws_size) return;

  auto cvt = [&](const float* src, h16* dst, int sr, int sc, int dc, int n) {
    k_cvt_pad<<<(n + 255) / 256, 256, 0, stream>>>(src, dst, sr, sc, dc, n);
  };
  auto cvtw = [&](int wIdx, int K, int N) {
    int n = N * K;
    k_cvt_tr<<<(n + 255) / 256, 256, 0, stream>>>((const float*)d_in[wIdx], wcvt,
                                                  K, N, K, n);
  };
  auto gemm = [&](const h16* Ap, int lda, int biasIdx, float* o32, h16* o16,
                  int M, int N, int K, int relu) {
    dim3 g(N / BN, M / BM);
    k_gemm<<<g, 256, 0, stream>>>(Ap, lda, wcvt, K, (const float*)d_in[biasIdx],
                                  o32, o16, N, K, relu);
  };
  auto attn = [&](int Lk) {
    k_vtr<<<dim3(Lk / 64, B_ * H_), 256, 0, stream>>>(v16, v16t, Lk);
    k_attn_wmma<<<dim3(L_ / 128, B_ * H_), 256, 0, stream>>>(q16, k16, v16t, ao16, Lk);
  };

  // ---- pos-embed MLP (K=120 padded to 128) ----
  cvt(vocab, vpad, L_, 120, 128, L_ * 128);
  { int n = F_ * 128;
    k_cvt_tr<<<(n + 255) / 256, 256, 0, stream>>>((const float*)d_in[100], wcvt,
                                                  120, F_, 128, n); }
  { dim3 g(F_ / BN, L_ / BM);
    k_gemm<<<g, 256, 0, stream>>>(vpad, 128, wcvt, 128, (const float*)d_in[99],
                                  nullptr, bigh, F_, 128, 1); }
  cvtw(102, F_, D_);
  gemm(bigh, F_, 101, tmp32, nullptr, L_, D_, F_, 0);
  k_bcast<<<(BL * D_ + 255) / 256, 256, 0, stream>>>(tmp32, xf32, xf16, BL * D_);

  cvt(bev, bev16, B_ * S_, D_, D_, B_ * S_ * D_);

  // ---- decoder layers ----
  for (int l = 0; l < 2; ++l) {
    const int lb = 47 + 26 * l;
    const int ca_k_b = lb + 0,  ca_k_w = lb + 1,  ca_o_b = lb + 2,  ca_o_w = lb + 3;
    const int ca_q_b = lb + 4,  ca_q_w = lb + 5,  ca_v_b = lb + 6,  ca_v_w = lb + 7;
    const int ff1_b  = lb + 8,  ff1_w  = lb + 9,  ff2_b  = lb + 10, ff2_w  = lb + 11;
    const int ln1_b  = lb + 12, ln1_g  = lb + 13, ln2_b  = lb + 14, ln2_g  = lb + 15;
    const int ln3_b  = lb + 16, ln3_g  = lb + 17;
    const int sa_k_b = lb + 18, sa_k_w = lb + 19, sa_o_b = lb + 20, sa_o_w = lb + 21;
    const int sa_q_b = lb + 22, sa_q_w = lb + 23, sa_v_b = lb + 24, sa_v_w = lb + 25;

    // self-attention
    cvtw(sa_q_w, D_, D_); gemm(xf16, D_, sa_q_b, nullptr, q16, BL, D_, D_, 0);
    cvtw(sa_k_w, D_, D_); gemm(xf16, D_, sa_k_b, nullptr, k16, BL, D_, D_, 0);
    cvtw(sa_v_w, D_, D_); gemm(xf16, D_, sa_v_b, nullptr, v16, BL, D_, D_, 0);
    attn(L_);
    cvtw(sa_o_w, D_, D_); gemm(ao16, D_, sa_o_b, tmp32, nullptr, BL, D_, D_, 0);
    k_ln<<<BL, 128, 0, stream>>>(xf32, tmp32, (const float*)d_in[ln1_g],
                                 (const float*)d_in[ln1_b], xf16);

    // cross-attention to BEV
    cvtw(ca_q_w, D_, D_); gemm(xf16, D_, ca_q_b, nullptr, q16, BL, D_, D_, 0);
    cvtw(ca_k_w, D_, D_); gemm(bev16, D_, ca_k_b, nullptr, k16, B_ * S_, D_, D_, 0);
    cvtw(ca_v_w, D_, D_); gemm(bev16, D_, ca_v_b, nullptr, v16, B_ * S_, D_, D_, 0);
    attn(S_);
    cvtw(ca_o_w, D_, D_); gemm(ao16, D_, ca_o_b, tmp32, nullptr, BL, D_, D_, 0);
    k_ln<<<BL, 128, 0, stream>>>(xf32, tmp32, (const float*)d_in[ln2_g],
                                 (const float*)d_in[ln2_b], xf16);

    // FFN
    cvtw(ff1_w, D_, F_); gemm(xf16, D_, ff1_b, nullptr, bigh, BL, F_, D_, 1);
    cvtw(ff2_w, F_, D_); gemm(bigh, F_, ff2_b, tmp32, nullptr, BL, D_, F_, 0);
    k_ln<<<BL, 128, 0, stream>>>(xf32, tmp32, (const float*)d_in[ln3_g],
                                 (const float*)d_in[ln3_b], xf16);
  }

  // dist = x + status
  k_addstatus<<<(BL * D_ + 255) / 256, 256, 0, stream>>>(xf32, status, xf16);

  // 2-layer heads (history_comfort skipped: unused by score formula)
  auto head2 = [&](int b0, int w0, int b1, int w1, int slot) {
    cvtw(w0, D_, F_);
    gemm(xf16, D_, b0, nullptr, bigh, BL, F_, D_, 1);
    k_matvec<<<BL / 8, 256, 0, stream>>>(bigh, (const float*)d_in[w1],
                                         (const float*)d_in[b1], rbuf + (size_t)slot * BL, F_);
  };
  head2(3, 4, 5, 6, 0);        // drivable_area_compliance
  head2(7, 8, 9, 10, 1);       // driving_direction_compliance
  head2(11, 12, 13, 14, 2);    // ego_progress
  head2(19, 20, 21, 22, 3);    // lane_keeping
  head2(23, 24, 25, 26, 4);    // no_at_fault_collisions
  head2(39, 40, 41, 42, 5);    // time_to_collision_within_bound
  head2(43, 44, 45, 46, 6);    // traffic_light_compliance

  auto head3 = [&](int base, int slot) {
    cvtw(base + 1, D_, F_);
    gemm(xf16, D_, base + 0, nullptr, bigh, BL, F_, D_, 1);
    cvtw(base + 3, F_, F_);
    gemm(bigh, F_, base + 2, nullptr, bigh2, BL, F_, F_, 1);
    k_matvec<<<BL / 8, 256, 0, stream>>>(bigh2, (const float*)d_in[base + 5],
                                         (const float*)d_in[base + 4],
                                         rbuf + (size_t)slot * BL, F_);
  };
  head3(27, 7);  // rl
  head3(33, 8);  // rl_topk

  k_scores<<<B_, 256, 0, stream>>>(rbuf, scores);
  k_topk<<<B_, 256, 0, stream>>>(scores, rbuf + (size_t)8 * BL);
  k_output<<<B_, 256, 0, stream>>>(scores, vocab, (float*)d_out);
}
